// Qwen3MoeSparseMoeBlock_2791728742951
// MI455X (gfx1250) — matrix operand order repaired and checked by reading, not compiled, not run on hardware
//
#include <hip/hip_runtime.h>
#include <hip/hip_bf16.h>
#include <stdint.h>

// Problem constants (match reference)
constexpr int T = 256;      // tokens
constexpr int H = 2048;     // hidden
constexpr int E = 64;       // experts
constexpr int I = 768;      // intermediate
constexpr int TOPK = 8;
constexpr int MT = 64;              // tokens per GEMM tile (4 x 16-row WMMA tiles)
constexpr int NMT = MT / 16;        // 4 M-tiles
constexpr int MAXTILES = T / MT;    // 4 (worst case: all tokens on one expert)

typedef __attribute__((ext_vector_type(16))) __bf16          v16bf;
typedef __attribute__((ext_vector_type(8)))  unsigned        v8u;
typedef __attribute__((ext_vector_type(8)))  float           v8f;

// ---- helpers -------------------------------------------------------------

__device__ __forceinline__ unsigned short f2bf(float f) {
    union { float f; unsigned u; } c; c.f = f;
    unsigned u = c.u + 0x7FFFu + ((c.u >> 16) & 1u);   // round-to-nearest-even
    return (unsigned short)(u >> 16);
}

// Round two f32 to bf16 (RNE) and pack into one dword with v_perm_b32:
// dst = { hi[31:16], lo[31:16] }  (5 VALU per 2 elements, no 16-bit movs)
__device__ __forceinline__ unsigned pack_bf16_pair(float flo, float fhi) {
    unsigned lo = __builtin_bit_cast(unsigned, flo);
    unsigned hi = __builtin_bit_cast(unsigned, fhi);
    lo += 0x7FFFu + ((lo >> 16) & 1u);
    hi += 0x7FFFu + ((hi >> 16) & 1u);
    return __builtin_amdgcn_perm(hi, lo, 0x07060302u);
}

// A-fragment (16x32 bf16): lane = row M (mod 16); lanes 0-15 hold K
// [k0..k0+15]; lanes 16-31 hold [k0+16..k0+31] (contiguous, matches B).
// Source rows are already bf16 (global or LDS); 16-byte aligned vector loads.
__device__ __forceinline__ v16bf load_a_bf16(const unsigned short* __restrict__ row,
                                             int k0, bool hi) {
    int b0 = k0 + (hi ? 16 : 0);
    union { uint4 q[2]; v16bf bf; } u;
    u.q[0] = *(const uint4*)(row + b0);
    u.q[1] = *(const uint4*)(row + b0 + 8);
    return u.bf;
}

// B-fragment (32x16 bf16, K x N): lane holds column N = lane%16, 16 contiguous
// K values starting at k0 (lanes 0-15) or k0+16 (lanes 16-31). Weight matrices
// are [N][K] row-major f32, so column N of B == weight row N (convert inline —
// weights are streamed from HBM exactly once).
__device__ __forceinline__ v16bf load_b_f32row(const float* __restrict__ wrow,
                                               int k0, bool hi) {
    int b0 = k0 + (hi ? 16 : 0);
    v8u v;
#pragma unroll
    for (int j = 0; j < 8; ++j)
        v[j] = pack_bf16_pair(wrow[b0 + 2 * j], wrow[b0 + 2 * j + 1]);
    return __builtin_bit_cast(v16bf, v);
}

__device__ __forceinline__ v8f wmma_bf16(v16bf a, v16bf b, v8f c) {
    return __builtin_amdgcn_wmma_f32_16x16x32_bf16(false, a, false, b,
                                                   (short)0, c, false, false);
}

// silu(g)*u with fast hardware reciprocal (avoids IEEE v_div_scale chains)
__device__ __forceinline__ float silu_mul(float g, float u) {
    float sig = __builtin_amdgcn_rcpf(1.f + __expf(-g));
    return g * sig * u;
}

// ---- kernel 0: one-shot X -> bf16 (removes all A-side conversions) -------

__global__ void x_to_bf16(const float* __restrict__ hs,
                          unsigned* __restrict__ xb) {   // dword view of bf16[T*H]
    int i = blockIdx.x * blockDim.x + threadIdx.x;       // indexes groups of 4
    float4 f = ((const float4*)hs)[i];
    uint2 o;
    o.x = pack_bf16_pair(f.x, f.y);
    o.y = pack_bf16_pair(f.z, f.w);
    ((uint2*)xb)[i] = o;
}

// ---- kernel 1: router (logits -> softmax -> top8 -> renorm) --------------

__global__ void moe_router(const float* __restrict__ hs,
                           const float* __restrict__ gw,
                           int* __restrict__ topk_idx,
                           float* __restrict__ topk_w) {
    int t = blockIdx.x;
    int e = threadIdx.x;                    // 64 threads, one per expert
    __shared__ float lg[E];

    const float* x = hs + (size_t)t * H;
    const float* w = gw + (size_t)e * H;
    float acc = 0.f;
#pragma unroll 4
    for (int k = 0; k < H; ++k) acc += x[k] * w[k];
    lg[e] = acc;
    __syncthreads();

    if (e == 0) {
        float m = lg[0];
        for (int i = 1; i < E; ++i) m = fmaxf(m, lg[i]);
        for (int i = 0; i < E; ++i) lg[i] = __expf(lg[i] - m);
        // top-8 of softmax numerators; renorm cancels the common denominator.
        bool used[E];
        for (int i = 0; i < E; ++i) used[i] = false;
        int   si[TOPK];
        float sp[TOPK];
        float wsum = 0.f;
        for (int k = 0; k < TOPK; ++k) {
            int   bj = -1;
            float bv = -1.f;
            for (int i = 0; i < E; ++i)
                if (!used[i] && lg[i] > bv) { bv = lg[i]; bj = i; }
            used[bj] = true; si[k] = bj; sp[k] = bv; wsum += bv;
        }
        float inv = 1.f / wsum;
        for (int k = 0; k < TOPK; ++k) {
            topk_idx[t * TOPK + k] = si[k];
            topk_w[t * TOPK + k]   = sp[k] * inv;
        }
    }
}

// ---- kernel 2: deterministic per-expert list build -----------------------

__global__ void moe_build(const int* __restrict__ topk_idx,
                          int* __restrict__ cnt,
                          int* __restrict__ off,
                          int* __restrict__ expert_tok,
                          int* __restrict__ row_of) {
    int e = threadIdx.x;                    // 64 threads, one per expert
    int c = 0;
    for (int s = 0; s < T * TOPK; ++s)
        if (topk_idx[s] == e) ++c;
    cnt[e] = c;
    __syncthreads();
    if (e == 0) {
        int s = 0;
        for (int i = 0; i < E; ++i) { off[i] = s; s += cnt[i]; }
    }
    __syncthreads();
    int o = off[e], pos = 0;
    for (int t = 0; t < T; ++t)
        for (int k = 0; k < TOPK; ++k) {
            int s = t * TOPK + k;
            if (topk_idx[s] == e) {
                expert_tok[o + pos] = t;
                row_of[s]           = o + pos;   // exactly one writer per slot
                ++pos;
            }
        }
}

// ---- kernel 3: grouped SwiGLU + down-proj GEMM (WMMA bf16) ---------------
// One block per (expert, 64-token tile): 4 M-tiles share every B fragment,
// so expert weights stream from HBM ~once (avg 32 tokens/expert -> 1 tile).
// 256 threads = 8 waves.

__global__ __launch_bounds__(256)
void moe_gemm(const unsigned short* __restrict__ xbf,   // [T,H] bf16
              const float* __restrict__ w13,
              const float* __restrict__ w2,
              const int* __restrict__ cnt,
              const int* __restrict__ off,
              const int* __restrict__ expert_tok,
              float* __restrict__ partial) {
    const int e    = blockIdx.x / MAXTILES;
    const int tile = blockIdx.x % MAXTILES;
    int rows = cnt[e] - tile * MT;          // block-uniform
    if (rows <= 0) return;                  // uniform exit: EXEC stays all-1s
    if (rows > MT) rows = MT;
    const int base = off[e] + tile * MT;    // first global partial-row

    __shared__ int            tokS[MT];
    __shared__ unsigned short actS[MT * I]; // 96 KB bf16 activations

    if (threadIdx.x < MT) {
        int r = threadIdx.x;
        tokS[r] = expert_tok[base + (r < rows ? r : 0)]; // pad w/ first token
    }
    __syncthreads();

    const int  wave = threadIdx.x >> 5;
    const int  lane = threadIdx.x & 31;
    const int  l16  = lane & 15;
    const bool hi   = lane >= 16;

    const unsigned short* xr[NMT];          // A row pointers (16 rows / M-tile)
#pragma unroll
    for (int t = 0; t < NMT; ++t)
        xr[t] = xbf + (size_t)tokS[t * 16 + l16] * H;

    // ---- phase 1: g = X*W1^T, u = X*W3^T, act = silu(g)*u -> LDS ----
    const size_t w13e = (size_t)e * 2 * I * H;
    for (int c = wave; c < I / 16; c += 8) {
        const int    n0 = c * 16;
        const float* wg = w13 + w13e + (size_t)(n0 + l16) * H;       // gate row
        const float* wu = wg + (size_t)I * H;                        // up row
        v8f g[NMT], u[NMT];
        const v8f vzero = {};
#pragma unroll
        for (int t = 0; t < NMT; ++t) { g[t] = vzero; u[t] = vzero; }
        for (int k = 0; k < H; k += 32) {
            v16bf bg = load_b_f32row(wg, k, hi);
            v16bf bu = load_b_f32row(wu, k, hi);
#pragma unroll
            for (int t = 0; t < NMT; ++t) {       // B reused across 4 M-tiles
                v16bf a = load_a_bf16(xr[t], k, hi);
                g[t] = wmma_bf16(a, bg, g[t]);
                u[t] = wmma_bf16(a, bu, u[t]);
            }
        }
        const int mh = hi ? 8 : 0;          // D layout: lanes16-31 hold M+8
#pragma unroll
        for (int t = 0; t < NMT; ++t)
#pragma unroll
            for (int r = 0; r < 8; ++r) {
                int m = t * 16 + r + mh;
                actS[m * I + n0 + l16] = f2bf(silu_mul(g[t][r], u[t][r]));
            }
    }
    __syncthreads();

    // ---- phase 2: partial = act @ W2^T  (unweighted down proj) ----
    const unsigned short* ar[NMT];
#pragma unroll
    for (int t = 0; t < NMT; ++t)
        ar[t] = actS + (t * 16 + l16) * I;

    const size_t w2e = (size_t)e * H * I;
    for (int c = wave; c < H / 16; c += 8) {
        const int    n0 = c * 16;
        const float* wr = w2 + w2e + (size_t)(n0 + l16) * I;
        v8f acc[NMT];
        const v8f vzero = {};
#pragma unroll
        for (int t = 0; t < NMT; ++t) acc[t] = vzero;
        for (int k = 0; k < I; k += 32) {
            v16bf b = load_b_f32row(wr, k, hi);
#pragma unroll
            for (int t = 0; t < NMT; ++t) {       // B reused across 4 M-tiles
                v16bf a = load_a_bf16(ar[t], k, hi);
                acc[t] = wmma_bf16(a, b, acc[t]);
            }
        }
        const int mh = hi ? 8 : 0;
#pragma unroll
        for (int t = 0; t < NMT; ++t)
#pragma unroll
            for (int r = 0; r < 8; ++r) {
                int m = t * 16 + r + mh;
                if (m < rows)
                    partial[(size_t)(base + m) * H + n0 + l16] = acc[t][r];
            }
    }
}

// ---- kernel 4: weighted combine (deterministic, no atomics) --------------

__global__ void moe_combine(const float* __restrict__ partial,
                            const float* __restrict__ topk_w,
                            const int* __restrict__ row_of,
                            float* __restrict__ out) {
    int t = blockIdx.x;
    int rowk[TOPK]; float wk[TOPK];
#pragma unroll
    for (int k = 0; k < TOPK; ++k) {
        rowk[k] = row_of[t * TOPK + k];
        wk[k]   = topk_w[t * TOPK + k];
    }
    for (int h = threadIdx.x; h < H; h += blockDim.x) {
        float s = 0.f;
#pragma unroll
        for (int k = 0; k < TOPK; ++k)
            s += wk[k] * partial[(size_t)rowk[k] * H + h];
        out[(size_t)t * H + h] = s;
    }
}

// ---- launch --------------------------------------------------------------

extern "C" void kernel_launch(void* const* d_in, const int* in_sizes, int n_in,
                              void* d_out, int out_size, void* d_ws, size_t ws_size,
                              hipStream_t stream) {
    const float* hs  = (const float*)d_in[0];   // [T,H]
    const float* gw  = (const float*)d_in[1];   // [E,H]
    const float* w13 = (const float*)d_in[2];   // [E,2I,H]
    const float* w2  = (const float*)d_in[3];   // [E,H,I]
    float* out = (float*)d_out;                 // [T,H]

    char* ws = (char*)d_ws;
    int*   topk_idx   = (int*)  (ws + 0);        // T*TOPK ints   (8 KB)
    float* topk_w     = (float*)(ws + 8192);     // T*TOPK floats (8 KB)
    int*   cnt        = (int*)  (ws + 16384);    // E
    int*   off        = (int*)  (ws + 16640);    // E
    int*   expert_tok = (int*)  (ws + 16896);    // T*TOPK        (8 KB)
    int*   row_of     = (int*)  (ws + 25088);    // T*TOPK        (8 KB)
    unsigned short* xbf = (unsigned short*)(ws + 65536);   // [T,H] bf16 (1 MB)
    float* partial    = (float*)(ws + 65536 + (size_t)T * H * 2); // 16.8 MB

    x_to_bf16  <<<(T * H / 4) / 256, 256, 0, stream>>>(hs, (unsigned*)xbf);
    moe_router <<<T, E,               0, stream>>>(hs, gw, topk_idx, topk_w);
    moe_build  <<<1, E,               0, stream>>>(topk_idx, cnt, off, expert_tok, row_of);
    moe_gemm   <<<E * MAXTILES, 256,  0, stream>>>(xbf, w13, w2, cnt, off, expert_tok, partial);
    moe_combine<<<T, 256,             0, stream>>>(partial, topk_w, row_of, out);
}
